// DynamicAggregation_28071906246674
// MI455X (gfx1250) — compile-verified, matrix operand order checked by reading, not run on hardware
//
#include <hip/hip_runtime.h>

// ---------------------------------------------------------------------------
// Types for CDNA5 WMMA (wave32): v_wmma_f32_16x16x32_bf16
// ---------------------------------------------------------------------------
typedef __attribute__((ext_vector_type(16))) __bf16 v16bf;
typedef __attribute__((ext_vector_type(8)))  float  v8f;

union BF16Frag {
    v16bf v;
    uint4 q[2];
    unsigned short us[16];
};

__device__ __forceinline__ float sigmoidf_(float x) { return 1.0f / (1.0f + __expf(-x)); }
__device__ __forceinline__ float siluf_(float x)    { return x * sigmoidf_(x); }

// float -> bf16 bits with round-to-nearest-even
__device__ __forceinline__ unsigned short f2bf(float f) {
    unsigned int u = __float_as_uint(f);
    u = (u + 0x7fffu + ((u >> 16) & 1u)) >> 16;
    return (unsigned short)u;
}
__device__ __forceinline__ unsigned int pack2bf(float a, float b) {
    return (unsigned int)f2bf(a) | ((unsigned int)f2bf(b) << 16);
}

// ---------------------------------------------------------------------------
// Generic bf16 WMMA GEMM:  Out[M,N] = act( A[M,K] @ Bm[K,N] + bias[N] )
// 64x64 tile per 128-thread (4-wave) workgroup, K streamed in 32-chunks.
// Requires: N multiple of 64, K multiple of 32 (true for all launches here).
// M may be ragged; only the last M-block takes the guarded slow path.
// act: 0 = none, 1 = SiLU.
// ---------------------------------------------------------------------------
#define LSTR 40  // LDS row stride in bf16 (80B: 16B-aligned, bank-spread)

__global__ __launch_bounds__(128) void gemm_bf16_wmma(
    const float* __restrict__ A, const float* __restrict__ Bm,
    float* __restrict__ Out, const float* __restrict__ bias,
    int M, int N, int K, int act)
{
    __shared__ unsigned short At[64 * LSTR];   // [m_local][k_local]
    __shared__ unsigned short Bt[64 * LSTR];   // [n_local][k_local] (transposed)

    const int tid  = threadIdx.x;
    const int lane = tid & 31;
    const int wave = tid >> 5;
    const int hi   = lane >> 4;   // K-half select
    const int ln   = lane & 15;
    const int wgM  = blockIdx.y * 64;
    const int wgN  = blockIdx.x * 64;
    const bool fullM = (wgM + 64 <= M);   // uniform across workgroup

    v8f zero = {0.f, 0.f, 0.f, 0.f, 0.f, 0.f, 0.f, 0.f};
    v8f acc[4];
    for (int mt = 0; mt < 4; ++mt) acc[mt] = zero;

    const int nK = K >> 5;
    for (int kt = 0; kt < nK; ++kt) {
        // ---- stage A tile 64x32 (f32 -> bf16) ----
        if (fullM) {
            // vectorized, unguarded: 4 iterations of float4 -> packed ds b64
            for (int i = tid; i < 512; i += 128) {       // 64*32/4 float4s
                int r = i >> 3, c4 = (i & 7) * 4;
                float4 v = *(const float4*)&A[(size_t)(wgM + r) * K + kt * 32 + c4];
                uint2 p; p.x = pack2bf(v.x, v.y); p.y = pack2bf(v.z, v.w);
                *(uint2*)&At[r * LSTR + c4] = p;
            }
        } else {
            for (int i = tid; i < 64 * 32; i += 128) {
                int r = i >> 5, c = i & 31;
                int gm = wgM + r;
                float v = (gm < M) ? A[(size_t)gm * K + (kt * 32 + c)] : 0.0f;
                At[r * LSTR + c] = f2bf(v);
            }
        }
        // ---- stage B tile 32x64 transposed -> Bt[n][k] (never ragged) ----
        for (int i = tid; i < 512; i += 128) {           // 32*64/4 float4s
            int r = i >> 4, c4 = (i & 15) * 4;           // r = k_local, c4 = n_local
            float4 v = *(const float4*)&Bm[(size_t)(kt * 32 + r) * N + wgN + c4];
            Bt[(c4 + 0) * LSTR + r] = f2bf(v.x);
            Bt[(c4 + 1) * LSTR + r] = f2bf(v.y);
            Bt[(c4 + 2) * LSTR + r] = f2bf(v.z);
            Bt[(c4 + 3) * LSTR + r] = f2bf(v.w);
        }
        // prefetch next k-tile while we compute (global_prefetch_b8)
        if (kt + 1 < nK) {
            int pr = tid >> 1, pc = (tid & 1) * 16;      // A: 64 rows x 2 chunks
            if (fullM || wgM + pr < M)
                __builtin_prefetch(&A[(size_t)(wgM + pr) * K + (kt + 1) * 32 + pc], 0, 3);
            int br = tid >> 2, bc = (tid & 3) * 16;      // B: 32 rows x 4 chunks
            __builtin_prefetch(&Bm[(size_t)((kt + 1) * 32 + br) * N + wgN + bc], 0, 3);
        }
        __syncthreads();

        // B fragment for this wave's n-slice: two 16B chunks (k=hi*8, k=hi*8+16)
        BF16Frag bf;
        const uint4* bp = (const uint4*)&Bt[(wave * 16 + ln) * LSTR + hi * 8];
        bf.q[0] = bp[0];
        bf.q[1] = bp[2];

        for (int mt = 0; mt < 4; ++mt) {
            BF16Frag af;
            const uint4* ap = (const uint4*)&At[(mt * 16 + ln) * LSTR + hi * 8];
            af.q[0] = ap[0];
            af.q[1] = ap[2];
            acc[mt] = __builtin_amdgcn_wmma_f32_16x16x32_bf16(
                false, af.v, false, bf.v, (short)0, acc[mt], false, false);
        }
        __syncthreads();
    }

    // ---- epilogue ----
    const int gn = wgN + wave * 16 + ln;
    float bn = bias ? bias[gn] : 0.0f;
    if (fullM) {
        for (int mt = 0; mt < 4; ++mt) {
            for (int r = 0; r < 8; ++r) {
                int gm = wgM + mt * 16 + r + hi * 8;
                float v = acc[mt][r] + bn;
                if (act) v = siluf_(v);
                Out[(size_t)gm * N + gn] = v;
            }
        }
    } else {
        for (int mt = 0; mt < 4; ++mt) {
            for (int r = 0; r < 8; ++r) {
                int gm = wgM + mt * 16 + r + hi * 8;
                if (gm < M) {
                    float v = acc[mt][r] + bn;
                    if (act) v = siluf_(v);
                    Out[(size_t)gm * N + gn] = v;
                }
            }
        }
    }
}

// ---------------------------------------------------------------------------
// omd[b] = 1 - sigmoid( dot(dh[b,:384], d2) + db2 )
// ---------------------------------------------------------------------------
__global__ __launch_bounds__(128) void difficulty_kernel(
    const float* __restrict__ dh, const float* __restrict__ d2,
    const float* __restrict__ db2, float* __restrict__ omd)
{
    __shared__ float red[128];
    int b = blockIdx.x;
    float s = 0.0f;
    for (int k = threadIdx.x; k < 384; k += 128) s += dh[(size_t)b * 384 + k] * d2[k];
    red[threadIdx.x] = s;
    __syncthreads();
    for (int off = 64; off > 0; off >>= 1) {
        if (threadIdx.x < off) red[threadIdx.x] += red[threadIdx.x + off];
        __syncthreads();
    }
    if (threadIdx.x == 0) omd[b] = 1.0f - sigmoidf_(red[0] + db2[0]);
}

// ---------------------------------------------------------------------------
// Pre-swizzle w2[384,192] (f32) into per-lane bf16 WMMA B-fragments:
// frag[((nt*12 + kt)*32 + lane)*16 + e], e -> k = kt*32 + (e&7) + (e>>3)*16 + hi*8
// One contiguous 32B block per lane => coalesced b128 loads in the hot loop.
// ---------------------------------------------------------------------------
__global__ void swizzle_w2(const float* __restrict__ w2, unsigned short* __restrict__ frag)
{
    int t = blockIdx.x * blockDim.x + threadIdx.x;
    if (t >= 12 * 12 * 32) return;
    int lane = t & 31;
    int kt   = (t >> 5) % 12;
    int nt   = t / (12 * 32);
    int hi   = lane >> 4, ln = lane & 15;
    int n    = nt * 16 + ln;

    __attribute__((aligned(16))) unsigned short out[16];
    for (int e = 0; e < 16; ++e) {
        int k = kt * 32 + (e & 7) + ((e >> 3) * 16) + hi * 8;
        out[e] = f2bf(w2[(size_t)k * 192 + n]);
    }
    uint4* dst = (uint4*)&frag[(size_t)t * 16];
    dst[0] = ((uint4*)out)[0];
    dst[1] = ((uint4*)out)[1];
}

// ---------------------------------------------------------------------------
// Fused per-task kernel (one workgroup = one task b, 4 waves):
//   h1[c,k] = silu( teproj[b,k] + ceproj[id,k] )            (b1 folded in teproj)
//   h2      = silu( h1 @ w2 + b2 )                           <- WMMA bf16
//   est     = sigmoid( h2 . w3 + b3 );  rel = sigmoid(est + cbias[id])
//   weight  = rel * omd[b] * valid; 2-bin histogram -> argmax label
// ---------------------------------------------------------------------------
#define H1STR 392   // bf16 row stride for h1 (784B: 16B-aligned, bank-spread)
#define H2STR 196   // f32 row stride for h2 partials

__global__ __launch_bounds__(128) void fused_task(
    const float* __restrict__ teproj, const float* __restrict__ ceproj,
    const unsigned short* __restrict__ w2frag,
    const float* __restrict__ b2, const float* __restrict__ w3,
    const float* __restrict__ b3,
    const int* __restrict__ ids, const int* __restrict__ labs,
    const float* __restrict__ cbias, const float* __restrict__ omd,
    float* __restrict__ out_labels, float* __restrict__ out_rel)
{
    union SMem {
        unsigned short h1[64 * H1STR];  // bf16 activations (layer-1 out)
        float h2p[64 * H2STR];          // f32 silu(h2)*w3 partials (reused)
    };
    __shared__ SMem sm;
    __shared__ float hist[2];

    const int b    = blockIdx.x;
    const int tid  = threadIdx.x;
    const int lane = tid & 31;
    const int wave = tid >> 5;
    const int hi   = lane >> 4;
    const int ln   = lane & 15;

    const float* tep   = teproj + (size_t)b * 384;
    const int*   myids = ids + (size_t)b * 64;

    // ---- build h1 in LDS (gather + add + silu + bf16) ----
    // each thread owns 3 fixed k-positions; hoist the task projection loads
    const float te0 = tep[tid], te1 = tep[tid + 128], te2 = tep[tid + 256];
    for (int c = 0; c < 64; ++c) {
        const float* cep = ceproj + (size_t)myids[c] * 384;
        sm.h1[c * H1STR + tid      ] = f2bf(siluf_(te0 + cep[tid]));
        sm.h1[c * H1STR + tid + 128] = f2bf(siluf_(te1 + cep[tid + 128]));
        sm.h1[c * H1STR + tid + 256] = f2bf(siluf_(te2 + cep[tid + 256]));
    }
    if (tid == 0) { hist[0] = 0.0f; hist[1] = 0.0f; }
    __syncthreads();

    // ---- layer 2: h1[64,384] @ w2[384,192] via WMMA bf16 ----
    // wave w owns n-tiles [3w, 3w+2]; each wave: 4 m-tiles x 3 n-tiles accums.
    v8f zero = {0.f, 0.f, 0.f, 0.f, 0.f, 0.f, 0.f, 0.f};
    v8f acc[4][3];
    for (int mt = 0; mt < 4; ++mt)
        for (int ntl = 0; ntl < 3; ++ntl) acc[mt][ntl] = zero;

    for (int kt = 0; kt < 12; ++kt) {
        BF16Frag af[4];
        for (int mt = 0; mt < 4; ++mt) {
            const uint4* ap = (const uint4*)&sm.h1[(mt * 16 + ln) * H1STR + kt * 32 + hi * 8];
            af[mt].q[0] = ap[0];
            af[mt].q[1] = ap[2];
        }
        for (int ntl = 0; ntl < 3; ++ntl) {
            int nt = wave * 3 + ntl;
            BF16Frag bf;
            const uint4* bp = (const uint4*)&w2frag[((size_t)(nt * 12 + kt) * 32 + lane) * 16];
            bf.q[0] = bp[0];
            bf.q[1] = bp[1];
            if (kt + 1 < 12)   // prefetch next k-tile of this n-panel
                __builtin_prefetch(&w2frag[((size_t)(nt * 12 + kt + 1) * 32 + lane) * 16], 0, 3);
            for (int mt = 0; mt < 4; ++mt) {
                acc[mt][ntl] = __builtin_amdgcn_wmma_f32_16x16x32_bf16(
                    false, af[mt].v, false, bf.v, (short)0, acc[mt][ntl], false, false);
            }
        }
    }
    __syncthreads();  // done reading h1; LDS is reused below

    // ---- epilogue: silu(h2 + b2) * w3 partials into LDS ----
    for (int mt = 0; mt < 4; ++mt) {
        for (int ntl = 0; ntl < 3; ++ntl) {
            int n = wave * 48 + ntl * 16 + ln;
            float bn = b2[n], wn = w3[n];
            for (int r = 0; r < 8; ++r) {
                int m = mt * 16 + r + hi * 8;
                sm.h2p[m * H2STR + n] = siluf_(acc[mt][ntl][r] + bn) * wn;
            }
        }
    }
    __syncthreads();

    // ---- layer 3 + double sigmoid + weighted 2-bin histogram ----
    if (tid < 64) {
        int c = tid;
        float s = 0.0f;
        for (int n = 0; n < 192; ++n) s += sm.h2p[c * H2STR + n];
        float est = sigmoidf_(s + b3[0]);
        int   id  = myids[c];
        float rel = sigmoidf_(est + cbias[id]);
        out_rel[(size_t)b * 64 + c] = rel;

        int lab = labs[(size_t)b * 64 + c];
        if (lab != -1) {
            float wt = rel * omd[b];
            atomicAdd(&hist[lab > 0 ? 1 : 0], wt);
        }
    }
    __syncthreads();
    if (tid == 0) out_labels[b] = (hist[1] > hist[0]) ? 1.0f : 0.0f;
}

// ---------------------------------------------------------------------------
// Host launcher
// ---------------------------------------------------------------------------
extern "C" void kernel_launch(void* const* d_in, const int* in_sizes, int n_in,
                              void* d_out, int out_size, void* d_ws, size_t ws_size,
                              hipStream_t stream)
{
    const float* te    = (const float*)d_in[0];   // [2048,768]
    const int*   ids   = (const int*)  d_in[1];   // [2048,64]
    const int*   labs  = (const int*)  d_in[2];   // [2048,64]
    const float* w1    = (const float*)d_in[3];   // [960,384]
    const float* b1    = (const float*)d_in[4];   // [384]
    const float* w2    = (const float*)d_in[5];   // [384,192]
    const float* b2    = (const float*)d_in[6];   // [192]
    const float* w3    = (const float*)d_in[7];   // [192,1]
    const float* b3    = (const float*)d_in[8];   // [1]
    const float* d1    = (const float*)d_in[9];   // [768,384]
    const float* db1   = (const float*)d_in[10];  // [384]
    const float* d2    = (const float*)d_in[11];  // [384,1]
    const float* db2   = (const float*)d_in[12];  // [1]
    const float* emb   = (const float*)d_in[13];  // [1000,192]
    const float* cbias = (const float*)d_in[14];  // [1000]

    char* ws = (char*)d_ws;
    float*          teproj = (float*)(ws + 0);                 // 2048*384 f32
    float*          dh     = (float*)(ws + 3145728);           // 2048*384 f32
    float*          ceproj = (float*)(ws + 6291456);           // 1024*384 f32 (1000 used)
    float*          omd    = (float*)(ws + 7864320);           // 2048 f32
    unsigned short* w2frag = (unsigned short*)(ws + 7872512);  // 12*12*32*16 bf16

    // teproj = TE @ w1[:768] + b1          (layer-1 task half, bias folded)
    gemm_bf16_wmma<<<dim3(6, 32), 128, 0, stream>>>(te, w1, teproj, b1, 2048, 384, 768, 0);
    // dh = silu(TE @ d1 + db1)             (difficulty hidden)
    gemm_bf16_wmma<<<dim3(6, 32), 128, 0, stream>>>(te, d1, dh, db1, 2048, 384, 768, 1);
    // ceproj = emb @ w1[768:]              (layer-1 contributor half, 1000 rows)
    gemm_bf16_wmma<<<dim3(6, 16), 128, 0, stream>>>(emb, w1 + (size_t)768 * 384, ceproj,
                                                    nullptr, 1000, 384, 192, 0);
    // omd[b] = 1 - sigmoid(dh[b].d2 + db2)
    difficulty_kernel<<<2048, 128, 0, stream>>>(dh, d2, db2, omd);
    // pre-swizzle w2 into WMMA B-fragments
    swizzle_w2<<<(12 * 12 * 32 + 127) / 128, 128, 0, stream>>>(w2, w2frag);
    // fused per-task layer-2/3 + histogram
    float* out_labels = (float*)d_out;       // [2048]
    float* out_rel    = out_labels + 2048;   // [2048*64]
    fused_task<<<2048, 128, 0, stream>>>(teproj, ceproj, w2frag, b2, w3, b3,
                                         ids, labs, cbias, omd, out_labels, out_rel);
}
